// SpotDetector_80470507258261
// MI455X (gfx1250) — matrix operand order repaired and checked by reading, not compiled, not run on hardware
//
#include <hip/hip_runtime.h>

// ---------------------------------------------------------------------------
// Spot detector for MI455X (gfx1250, wave32, WMMA, async global->LDS).
//
//   k_init        : zero candidate counter
//   k_convert     : image f32 -> f16
//   k_packpsf     : pack PSF into per-lane WMMA B-fragment records (32B/lane)
//   k_rowprefix   : per-row inclusive prefix sums (disk conv)
//   k_bgconv      : disk background conv via prefix-sum interval sums
//   k_wmma        : implicit-GEMM PSF correlation. Block = 16px x 4rows,
//                   4 waves (one y-row each). Stages 36 image rows (3.4KB)
//                   + full packed PSF-B (32KB, global_load_async_to_lds_b128)
//                   into LDS, then a 32-step K-loop of pure LDS + WMMA.
//                   Fused ratio + max-over-z (16-lane shfl_xor) epilogue.
//   k_rowmax/k_collect : separable 32x32 local-max + candidate append
//   k_topk        : single-workgroup bitonic top-1024 (LDS)
//   k_rois        : 32x32 ROI gather
// ---------------------------------------------------------------------------

#define HH 1024
#define WW 1024
#define ZP 16
#define PH 32
#define PW 32
#define THRESH 2.0f
#define KTOP 1024
#define ROISZ 32
#define BGR 64
#define CAP 4096

typedef _Float16 v16h __attribute__((ext_vector_type(16)));
typedef _Float16 v8h  __attribute__((ext_vector_type(8)));
typedef float    v8f  __attribute__((ext_vector_type(8)));

// ---------------- workspace layout (byte offsets) ----------------
#define WS_IMGF16    0x0000000   // 1024*1024 f16      = 2 MB
#define WS_ROWPS     0x0200000   // 1024*1024 f32      = 4 MB
#define WS_BG        0x0600000   // 1024*1024 f32      = 4 MB
#define WS_RESULT    0x0A00000   // 1024*1024 f32      = 4 MB
#define WS_ROWMAX    0x0E00000   // 1024*1024 f32      = 4 MB
#define WS_PSFB      0x1200000   // 32*32*16 f16       = 32 KB (16B aligned)
#define WS_CANDV     0x1210000   // CAP f32
#define WS_CANDI     0x1214000   // CAP i32
#define WS_CNT       0x1218000   // 1 i32
#define WS_SROY      0x1219000   // 1024 i32
#define WS_SROX      0x121A000   // 1024 i32
#define WS_SVAL      0x121B000   // 1024 f32

__global__ void k_init(int* cnt) { if (threadIdx.x == 0) *cnt = 0; }

__global__ void k_convert(const float* __restrict__ img, _Float16* __restrict__ out) {
    int id = blockIdx.x * blockDim.x + threadIdx.x;
    if (id < HH * WW) out[id] = (_Float16)img[id];
}

// Pack psf[z][dy][dx] into B-fragment layout for v_wmma_f32_16x16x32_f16.
// B is K(32) x N(16); wave32 layout: lanes 0-15 hold K=0..7,16..23 (VGPR v ->
// K=2v,2v+1 / 16+2(v-4).. for v>=4), lanes 16-31 hold those K+8.  N (= z) = lane & 15.  Stored as
// psfB[dy][lane][16 halves] = one aligned 32B record per lane per dy.
__global__ void k_packpsf(const float* __restrict__ psf, _Float16* __restrict__ psfB) {
    int id = blockIdx.x * blockDim.x + threadIdx.x;   // 32*256 = 8192 threads
    if (id >= 32 * 256) return;
    int dy   = id >> 8;
    int r    = id & 255;
    int v    = r >> 5;
    int lane = r & 31;
    int z    = lane & 15;
    int k0   = 2 * v + ((lane < 16) ? 0 : 8) + ((v >= 4) ? 8 : 0);
    const float* src = psf + (z * PH + dy) * PW;
    _Float16* dst = psfB + ((size_t)dy * 32 + lane) * 16 + 2 * v;
    dst[0] = (_Float16)src[k0];
    dst[1] = (_Float16)src[k0 + 1];
}

// One wave per row; sequential 32-wide chunks with wave scan.
__global__ void k_rowprefix(const float* __restrict__ img, float* __restrict__ ps) {
    int row  = blockIdx.x;
    int lane = threadIdx.x;
    const float* r = img + (size_t)row * WW;
    float* o = ps + (size_t)row * WW;
    float carry = 0.0f;
    for (int c = 0; c < WW; c += 32) {
        float v = r[c + lane];
        for (int d = 1; d < 32; d <<= 1) {
            float t = __shfl_up(v, d, 32);
            if (lane >= d) v += t;
        }
        o[c + lane] = carry + v;
        carry += __shfl(v, 31, 32);
    }
}

// Disk conv (circular, matching FFT wraparound) via per-row interval sums.
__global__ void k_bgconv(const float* __restrict__ ps, float* __restrict__ bg) {
    int id = blockIdx.x * blockDim.x + threadIdx.x;
    if (id >= HH * WW) return;
    int x = id & (WW - 1), y = id >> 10;
    float s = 0.0f;
    for (int dy = -(BGR - 1); dy <= (BGR - 1); ++dy) {
        int w  = (int)sqrtf((float)(BGR * BGR - 1 - dy * dy));
        int ry = (y + dy) & (HH - 1);
        const float* p = ps + (size_t)ry * WW;
        int a = x - w, b = x + w;
        float seg;
        if (a < 0)              seg = p[b] + (p[WW - 1] - p[a + WW - 1]);
        else if (b > WW - 1)    seg = (p[WW - 1] - (a ? p[a - 1] : 0.0f)) + p[b - WW];
        else                    seg = p[b] - (a ? p[a - 1] : 0.0f);
        s += seg;
    }
    bg[id] = s * (1.0f / (3.14159265358979f * (float)(BGR * BGR)));
}

// Implicit-GEMM PSF correlation.
// Block: 128 threads = 4 waves; wave w computes 16 pixels at row y0+w, all 16
// z-planes.  LDS holds the full packed PSF-B (32KB) + 36 staged image rows.
__global__ __launch_bounds__(128) void k_wmma(const _Float16* __restrict__ img,
                                              const _Float16* __restrict__ psfB,
                                              const float* __restrict__ bg,
                                              float* __restrict__ result) {
    __shared__ alignas(16) _Float16 sB[PH * 32 * 16];   // 32 KB packed B
    __shared__ _Float16 sRow[PH + 4][48];               // 36 rows x 48 halves

    const int lane = threadIdx.x & 31;
    const int wv   = threadIdx.x >> 5;
    const int x0   = blockIdx.x * 16;
    const int y0   = blockIdx.y * 4;

    // ---- stage packed PSF-B via async global->LDS (ASYNCcnt-tracked) ----
    {
        unsigned ldsbase = (unsigned)(uintptr_t)(&sB[0]);
        unsigned long long gbase = (unsigned long long)(uintptr_t)psfB;
        for (int i = threadIdx.x; i < (PH * 32 * 16 * 2) / 16; i += 128) {
            unsigned laddr = ldsbase + (unsigned)i * 16u;
            unsigned long long gaddr = gbase + (unsigned long long)i * 16ull;
            asm volatile("global_load_async_to_lds_b128 %0, %1, off"
                         :: "v"(laddr), "v"(gaddr) : "memory");
        }
    }
    // ---- stage the 36 wrapped image rows this block needs ----
    __builtin_prefetch(bg + (size_t)y0 * WW + x0, 0, 1);
    for (int i = threadIdx.x; i < 36 * 48; i += 128) {
        int r = i / 48, c = i - r * 48;
        int gy = (y0 + r - PH / 2) & (HH - 1);
        int gx = (x0 - PW / 2 + c) & (WW - 1);
        sRow[r][c] = img[(size_t)gy * WW + gx];
    }
    asm volatile("s_wait_asynccnt 0x0" ::: "memory");
    __syncthreads();

    // ---- K-loop: 32 x v_wmma_f32_16x16x32_f16, pure LDS operands ----
    const int M   = lane & 15;
    const int hi8 = (lane >= 16) ? 8 : 0;   // hi half-wave: K-list is lo+8
    v8f acc = {};
    for (int dy = 0; dy < PH; ++dy) {
        // A fragment: two contiguous 8-half runs at base+0..7 and base+16..23.
        const _Float16* ap = &sRow[wv + dy][M + hi8];
        v16h a;
        #pragma unroll
        for (int i = 0; i < 8; ++i) { a[i] = ap[i]; a[8 + i] = ap[16 + i]; }
        // B fragment: one aligned 32B record -> two ds_load_b128.
        const v8h* bp = (const v8h*)(sB + ((size_t)dy * 32 + lane) * 16);
        v8h b0 = bp[0], b1 = bp[1];
        v16h b;
        #pragma unroll
        for (int i = 0; i < 8; ++i) { b[i] = b0[i]; b[8 + i] = b1[i]; }
        acc = __builtin_amdgcn_wmma_f32_16x16x32_f16(false, a, false, b,
                                                     (short)0, acc, false, false);
    }

    // ---- epilogue: ratio = conv/(bg+1); max over z across 16-lane group ----
    const int y = y0 + wv;
    for (int r = 0; r < 8; ++r) {
        const int m  = (lane < 16) ? r : r + 8;   // D VGPR r -> pixel m
        const int px = x0 + m;
        const float bgv = bg[(size_t)y * WW + px];
        float ratio = acc[r] / (bgv + 1.0f);
        for (int msk = 1; msk < 16; msk <<= 1)
            ratio = fmaxf(ratio, __shfl_xor(ratio, msk, 32));
        if ((lane & 15) == 0) result[(size_t)y * WW + px] = ratio;
    }
}

// Separable 32-window max-pool, pass 1: row max over [x-16, x+15].
__global__ void k_rowmax(const float* __restrict__ result, float* __restrict__ rm) {
    int id = blockIdx.x * blockDim.x + threadIdx.x;
    if (id >= HH * WW) return;
    int x = id & (WW - 1), y = id >> 10;
    int lo = x - 16 > 0 ? x - 16 : 0;
    int hi = x + 15 < WW - 1 ? x + 15 : WW - 1;
    const float* rr = result + (size_t)y * WW;
    float m = -__builtin_inff();
    for (int b = lo; b <= hi; ++b) m = fmaxf(m, rr[b]);
    rm[id] = m;
}

// Pass 2: col max + local-max mask + candidate collection.
__global__ void k_collect(const float* __restrict__ result, const float* __restrict__ rm,
                          int* __restrict__ cnt, float* __restrict__ cv, int* __restrict__ ci) {
    int id = blockIdx.x * blockDim.x + threadIdx.x;
    if (id >= HH * WW) return;
    int x = id & (WW - 1), y = id >> 10;
    float r = result[id];
    float mx = 0.0f;                           // top-left zero-pad row/col
    if (x != 0 && y != 0) {
        int lo = y - 16 > 0 ? y - 16 : 0;
        int hi = y + 15 < HH - 1 ? y + 15 : HH - 1;
        mx = -__builtin_inff();
        for (int a = lo; a <= hi; ++a) mx = fmaxf(mx, rm[(size_t)a * WW + x]);
    }
    if (mx == r && r > THRESH) {
        int p = atomicAdd(cnt, 1);
        if (p < CAP) { cv[p] = r; ci[p] = id; }
    }
}

__device__ inline bool worse(float av, int ai, float bv, int bi) {
    return (av < bv) || (av == bv && ai > bi);
}

// Single-workgroup bitonic top-K over <= CAP candidates (zero-padded).
__global__ __launch_bounds__(1024) void k_topk(const int* __restrict__ cnt,
                                               const float* __restrict__ cv,
                                               const int* __restrict__ ci,
                                               float* __restrict__ out,
                                               int* __restrict__ sroy,
                                               int* __restrict__ srox,
                                               float* __restrict__ sval) {
    __shared__ float sv[CAP];
    __shared__ int   si[CAP];
    int n = *cnt; if (n > CAP) n = CAP;
    for (int i = threadIdx.x; i < CAP; i += blockDim.x) {
        if (i < n) { sv[i] = cv[i]; si[i] = ci[i]; }
        else       { sv[i] = 0.0f;  si[i] = 0; }
    }
    __syncthreads();
    for (int k = 2; k <= CAP; k <<= 1) {
        for (int j = k >> 1; j > 0; j >>= 1) {
            for (int i = threadIdx.x; i < CAP; i += blockDim.x) {
                int ixj = i ^ j;
                if (ixj > i) {
                    bool descBlock = ((i & k) == 0);
                    float v1 = sv[i], v2 = sv[ixj];
                    int   i1 = si[i], i2 = si[ixj];
                    bool sw = descBlock ? worse(v1, i1, v2, i2)
                                        : worse(v2, i2, v1, i1);
                    if (sw) { sv[i] = v2; si[i] = i2; sv[ixj] = v1; si[ixj] = i1; }
                }
            }
            __syncthreads();
        }
    }
    int t = threadIdx.x;                       // blockDim == KTOP
    float val = sv[t];
    int idx = si[t];
    int x = idx & (WW - 1), y = idx >> 10;
    int roy = y - ROISZ / 2, rox = x - ROISZ / 2;
    bool valid = (val > THRESH) && roy >= 0 && roy < (HH - ROISZ)
                                && rox >= 0 && rox < (WW - ROISZ);
    int royc = roy < 0 ? 0 : (roy > HH - ROISZ ? HH - ROISZ : roy);
    int roxc = rox < 0 ? 0 : (rox > WW - ROISZ ? WW - ROISZ : rox);
    float vm = valid ? 1.0f : 0.0f;
    out[2 * t + 0] = (float)royc * vm;                       // roipos
    out[2 * t + 1] = (float)roxc * vm;
    out[2 * KTOP + t] = val * vm;                            // intensity
    out[3 * KTOP + (size_t)KTOP * ROISZ * ROISZ + t] = vm;   // valid
    sroy[t] = royc; srox[t] = roxc; sval[t] = vm;
}

// 32x32 ROI gather per spot, masked by valid.
__global__ void k_rois(const float* __restrict__ img,
                       const int* __restrict__ sroy, const int* __restrict__ srox,
                       const float* __restrict__ sval, float* __restrict__ out) {
    int s = blockIdx.x;
    int roy = sroy[s], rox = srox[s];
    float vm = sval[s];
    float* o = out + 3 * KTOP + (size_t)s * ROISZ * ROISZ;
    for (int t = threadIdx.x; t < ROISZ * ROISZ; t += blockDim.x) {
        int ry = t >> 5, rx = t & 31;
        o[t] = img[(size_t)(roy + ry) * WW + (rox + rx)] * vm;
    }
}

extern "C" void kernel_launch(void* const* d_in, const int* in_sizes, int n_in,
                              void* d_out, int out_size, void* d_ws, size_t ws_size,
                              hipStream_t stream) {
    const float* image = (const float*)d_in[0];
    const float* psf   = (const float*)d_in[1];
    float* out = (float*)d_out;

    char* ws = (char*)d_ws;
    _Float16* imgF16 = (_Float16*)(ws + WS_IMGF16);
    float*    rowps  = (float*)(ws + WS_ROWPS);
    float*    bg     = (float*)(ws + WS_BG);
    float*    result = (float*)(ws + WS_RESULT);
    float*    rowmax = (float*)(ws + WS_ROWMAX);
    _Float16* psfB   = (_Float16*)(ws + WS_PSFB);
    float*    candV  = (float*)(ws + WS_CANDV);
    int*      candI  = (int*)(ws + WS_CANDI);
    int*      cnt    = (int*)(ws + WS_CNT);
    int*      sroy   = (int*)(ws + WS_SROY);
    int*      srox   = (int*)(ws + WS_SROX);
    float*    sval   = (float*)(ws + WS_SVAL);

    k_init<<<1, 32, 0, stream>>>(cnt);
    k_convert<<<(HH * WW + 255) / 256, 256, 0, stream>>>(image, imgF16);
    k_packpsf<<<32, 256, 0, stream>>>(psf, psfB);
    k_rowprefix<<<HH, 32, 0, stream>>>(image, rowps);
    k_bgconv<<<(HH * WW + 255) / 256, 256, 0, stream>>>(rowps, bg);
    k_wmma<<<dim3(WW / 16, HH / 4), 128, 0, stream>>>(imgF16, psfB, bg, result);
    k_rowmax<<<(HH * WW + 255) / 256, 256, 0, stream>>>(result, rowmax);
    k_collect<<<(HH * WW + 255) / 256, 256, 0, stream>>>(result, rowmax, cnt, candV, candI);
    k_topk<<<1, 1024, 0, stream>>>(cnt, candV, candI, out, sroy, srox, sval);
    k_rois<<<KTOP, 256, 0, stream>>>(image, sroy, srox, sval, out);
}
